// DimeNetPP_28587302322454
// MI455X (gfx1250) — matrix operand order repaired and checked by reading, not compiled, not run on hardware
//
#include <hip/hip_runtime.h>
#include <hip/hip_bf16.h>

#define CUTOFF 5.0f
#define NA 512
#define HID 128
#define NRBF 60
#define NB 4
#define NMOLS 16
#define NCHUNK 8   // i-dimension split of the msg reduction (occupancy)

typedef __attribute__((ext_vector_type(16))) _Float16 v16h;
typedef __attribute__((ext_vector_type(8)))  _Float16 v8h;
typedef __attribute__((ext_vector_type(8)))  float    v8f;

__device__ __forceinline__ v8f wmma_f16(v16h a, v16h b, v8f c) {
  // D = A(16x32 f16) x B(32x16 f16) + C(16x16 f32)
  return __builtin_amdgcn_wmma_f32_16x16x32_f16(
      /*neg_a=*/false, a, /*neg_b=*/false, b,
      /*c_mod=*/(short)0, c, /*reuse_a=*/false, /*reuse_b=*/false);
}

// A fragment from row-major f16 (LDS), per CDNA5 16-bit A layout:
// lane<16: row=lane, K = kstep+{0..7} (elems 0-7), kstep+{16..23} (elems 8-15)
// lane>=16: row=lane-16, K = kstep+{8..15}, kstep+{24..31}
__device__ __forceinline__ v16h load_A(const _Float16* base, int lane, int stride, int kstep) {
  const _Float16* p = base + (lane & 15) * stride + kstep + ((lane >> 4) << 3);
  v8h lo = *(const v8h*)p;
  v8h hi = *(const v8h*)(p + 16);
  v16h f;
#pragma unroll
  for (int e = 0; e < 8; ++e) { f[e] = lo[e]; f[e + 8] = hi[e]; }
  return f;
}

// B fragment from pre-packed fragment-major weights:
// packed[((frag)*32 + lane)*16 + e]  ->  two contiguous 16B loads per lane.
__device__ __forceinline__ v16h load_Bp(const _Float16* Bp, int frag, int lane) {
  const _Float16* p = Bp + (frag * 32 + lane) * 16;
  v8h lo = *(const v8h*)p;
  v8h hi = *(const v8h*)(p + 8);
  v16h f;
#pragma unroll
  for (int e = 0; e < 8; ++e) { f[e] = lo[e]; f[e + 8] = hi[e]; }
  return f;
}

__device__ __forceinline__ float silu(float z) {
  return z / (1.0f + __expf(-z));
}

// Map packed fragment index -> (k, col) of a KxHID row-major weight.
// frag = ntile*nk + ks ; col = ntile*16 + (lane&15) ; k = ks*32 + (lane>=16?8:0) + (e&7) + (e>=8?16:0)
__device__ __forceinline__ void unpack_idx(int pidx, int nk, int* k, int* col) {
  int e = pidx & 15;
  int lane = (pidx >> 4) & 31;
  int frag = pidx >> 9;
  int ks = frag % nk, nt = frag / nk;
  *col = nt * 16 + (lane & 15);
  *k = ks * 32 + ((lane >> 4) << 3) + (e & 7) + ((e >> 3) << 4);
}

// ---------------------------------------------------------------------------
// Kernel 1: embedding gather + weight f16 conversion, packed into WMMA
// B-fragment layout (w1r zero-padded 60->64 K rows).
// ---------------------------------------------------------------------------
__global__ void prep_kernel(const int* an, const float* emb,
                            const float* msg_w1, const float* msg_w2,
                            const float* upd_w1, const float* upd_w2,
                            float* x, _Float16* xh,
                            _Float16* w1x_p, _Float16* w1r_p, _Float16* w2_p,
                            _Float16* u1_p, _Float16* u2_p) {
  const int NX = NA * HID;            // 65536
  const int NW1X = NB * HID * HID;    // 65536   (nk=4)
  const int NW1R = NB * 64 * HID;     // 32768   (nk=2)
  const int NW2 = NB * HID * HID;     // 65536   (nk=4)
  const int NU1 = NB * 2 * HID * HID; // 131072  (nk=8)
  const int NU2 = NB * HID * HID;     // 65536   (nk=4)
  int total = NX + NW1X + NW1R + NW2 + NU1 + NU2;
  for (int idx = blockIdx.x * blockDim.x + threadIdx.x; idx < total;
       idx += gridDim.x * blockDim.x) {
    int i = idx;
    if (i < NX) {
      int a = i >> 7, c = i & 127;
      int z = an[a]; z = z < 0 ? 0 : (z > 99 ? 99 : z);
      float v = emb[z * HID + c];
      x[i] = v; xh[i] = (_Float16)v;
      continue;
    }
    i -= NX;
    if (i < NW1X) {
      int b = i / (HID * HID), p = i % (HID * HID), k, col;
      unpack_idx(p, 4, &k, &col);
      w1x_p[i] = (_Float16)msg_w1[(b * (HID + NRBF) + k) * HID + col];
      continue;
    }
    i -= NW1X;
    if (i < NW1R) {
      int b = i / (64 * HID), p = i % (64 * HID), k, col;
      unpack_idx(p, 2, &k, &col);
      float v = (k < NRBF) ? msg_w1[(b * (HID + NRBF) + HID + k) * HID + col] : 0.0f;
      w1r_p[i] = (_Float16)v;
      continue;
    }
    i -= NW1R;
    if (i < NW2) {
      int b = i / (HID * HID), p = i % (HID * HID), k, col;
      unpack_idx(p, 4, &k, &col);
      w2_p[i] = (_Float16)msg_w2[b * HID * HID + k * HID + col];
      continue;
    }
    i -= NW2;
    if (i < NU1) {
      int b = i / (2 * HID * HID), p = i % (2 * HID * HID), k, col;
      unpack_idx(p, 8, &k, &col);
      u1_p[i] = (_Float16)upd_w1[b * 2 * HID * HID + k * HID + col];
      continue;
    }
    i -= NU1;
    {
      int b = i / (HID * HID), p = i % (HID * HID), k, col;
      unpack_idx(p, 4, &k, &col);
      u2_p[i] = (_Float16)upd_w2[b * HID * HID + k * HID + col];
    }
  }
}

// ---------------------------------------------------------------------------
// Kernel 2: T = x @ w1x + b1  (512x128 @ 128x128), WMMA
// grid 32 (16-row tiles), 8 waves = 8 column tiles of 16
// ---------------------------------------------------------------------------
__global__ __launch_bounds__(256) void tgemm_kernel(const _Float16* __restrict__ xh,
                                                    const _Float16* __restrict__ w1x_p,
                                                    const float* __restrict__ b1,
                                                    float* __restrict__ T) {
  __shared__ __align__(16) _Float16 sX[16 * HID];
  int tid = threadIdx.x, wave = tid >> 5, lane = tid & 31;
  int rbase = blockIdx.x * 16;
  for (int idx = tid; idx < 16 * HID; idx += 256) sX[idx] = xh[rbase * HID + idx];
  __syncthreads();
  v8f acc = {};
#pragma unroll
  for (int ks = 0; ks < 4; ++ks)
    acc = wmma_f16(load_A(sX, lane, HID, ks * 32),
                   load_Bp(w1x_p, wave * 4 + ks, lane), acc);
  int col = wave * 16 + (lane & 15);
  int mbase = (lane < 16) ? 0 : 8;
  float bb = b1[col];
#pragma unroll
  for (int v = 0; v < 8; ++v)
    T[(rbase + mbase + v) * HID + col] = acc[v] + bb;
}

// ---------------------------------------------------------------------------
// Kernel 3 (hot): fused dist -> RBF(f16,LDS) -> WMMA(16x16x32 f16 x2)
// -> +T -> SiLU -> mask -> partial reduce over an i-chunk.
// grid (32 j-tiles, NCHUNK i-chunks); 8 waves = 8 h-tiles; 4 i-tiles per WG.
// Writes partial aggr/deg (deterministically reduced by reduce_kernel).
// ---------------------------------------------------------------------------
__global__ __launch_bounds__(256) void msg_kernel(const float* __restrict__ pos,
                                                  const float* __restrict__ centers,
                                                  const float* __restrict__ widths,
                                                  const float* __restrict__ T,
                                                  const _Float16* __restrict__ w1r_p,
                                                  float* __restrict__ aggrP,
                                                  float* __restrict__ degP) {
  __shared__ __align__(16) _Float16 sA[256 * 64]; // 256 pairs x 64 K (f16)
  __shared__ float sMask[256];
  __shared__ float sT[16 * HID];
  __shared__ float sDeg[16];
  __shared__ float sPosJ[48];
  __shared__ float sCent[64];
  __shared__ float sInvW[64];

  int tid = threadIdx.x, wave = tid >> 5, lane = tid & 31;
  int jbase = blockIdx.x * 16;
  int chunk = blockIdx.y;
  if (tid < 48) sPosJ[tid] = pos[jbase * 3 + tid];
  if (tid < 16) sDeg[tid] = 0.0f;
  if (tid < 64) {
    if (tid < NRBF) {
      sCent[tid] = centers[tid];
      float w = widths[tid];
      sInvW[tid] = 0.5f / (w * w);
    } else { sCent[tid] = 0.0f; sInvW[tid] = 0.0f; }
  }
  __syncthreads();

  // weight fragments: register-resident for the whole kernel (2x16B per lane)
  v16h bf0 = load_Bp(w1r_p, wave * 2 + 0, lane);
  v16h bf1 = load_Bp(w1r_p, wave * 2 + 1, lane);

  int il = tid >> 4, jl = tid & 15;
  float pjx = sPosJ[jl * 3 + 0], pjy = sPosJ[jl * 3 + 1], pjz = sPosJ[jl * 3 + 2];
  int hcol = wave * 16 + (lane & 15);
  int mbase = (lane < 16) ? 0 : 8;
  v8f acc = {}; // partial aggr[16 j x 16 h] tile in D layout

  int it0 = chunk * (32 / NCHUNK), it1 = it0 + (32 / NCHUNK);
  for (int it = it0; it < it1; ++it) {
    int ibase = it * 16;
    // ---- stage: distances, mask, RBF (f16), T tile ----
    {
      int i = ibase + il;
      float dx = pos[i * 3 + 0] - pjx;
      float dy = pos[i * 3 + 1] - pjy;
      float dz = pos[i * 3 + 2] - pjz;
      float d2 = dx * dx + dy * dy + dz * dz;
      bool eye = (i == jbase + jl);
      float dist = sqrtf(eye ? 1.0f : d2);
      float mk = (!eye && dist < CUTOFF) ? 1.0f : 0.0f;
      sMask[tid] = mk;
      atomicAdd(&sDeg[jl], mk); // LDS-only; reduced deterministically later
      _Float16* ap = &sA[tid * 64];
#pragma unroll
      for (int r = 0; r < 64; ++r) {
        float dd = dist - sCent[r];
        float e = __expf(-dd * dd * sInvW[r]);
        ap[r] = (_Float16)((r < NRBF) ? e : 0.0f);
      }
#pragma unroll
      for (int e = 0; e < 8; ++e)
        sT[tid * 8 + e] = T[ibase * HID + tid * 8 + e];
    }
    __syncthreads();
    // ---- compute: 16 m-tiles (one per i), 2 k-steps each ----
    for (int mt = 0; mt < 16; ++mt) {
      v8f d = {};
      d = wmma_f16(load_A(&sA[mt * 16 * 64], lane, 64, 0), bf0, d);
      d = wmma_f16(load_A(&sA[mt * 16 * 64], lane, 64, 32), bf1, d);
      float tval = sT[mt * HID + hcol]; // T already includes msg_b1
#pragma unroll
      for (int v = 0; v < 8; ++v) {
        float mk = sMask[mt * 16 + mbase + v];
        acc[v] += silu(d[v] + tval) * mk;
      }
    }
    __syncthreads();
  }
#pragma unroll
  for (int v = 0; v < 8; ++v)
    aggrP[chunk * (NA * HID) + (jbase + mbase + v) * HID + hcol] = acc[v];
  if (tid < 16) degP[chunk * NA + jbase + tid] = sDeg[tid];
}

// ---------------------------------------------------------------------------
// Kernel 3b: deterministic reduction of msg partials (fixed summation order)
// ---------------------------------------------------------------------------
__global__ __launch_bounds__(256) void reduce_kernel(const float* __restrict__ aggrP,
                                                     const float* __restrict__ degP,
                                                     float* __restrict__ aggr,
                                                     float* __restrict__ deg) {
  int total = NA * HID + NA;
  for (int idx = blockIdx.x * blockDim.x + threadIdx.x; idx < total;
       idx += gridDim.x * blockDim.x) {
    if (idx < NA * HID) {
      float s = 0.0f;
#pragma unroll
      for (int c = 0; c < NCHUNK; ++c) s += aggrP[c * (NA * HID) + idx];
      aggr[idx] = s;
    } else {
      int j = idx - NA * HID;
      float s = 0.0f;
#pragma unroll
      for (int c = 0; c < NCHUNK; ++c) s += degP[c * NA + j];
      deg[j] = s;
    }
  }
}

// ---------------------------------------------------------------------------
// Kernel 4: fused update — m2 = aggr@w2 + deg*b2 ; g = silu([x|m2]@upd_w1 + b1)
//           x += g@upd_w2 + b2.  Three WMMA GEMM stages, one 16-row tile per WG.
// ---------------------------------------------------------------------------
__global__ __launch_bounds__(256) void upd_kernel(float* __restrict__ x,
                                                  _Float16* __restrict__ xh,
                                                  const float* __restrict__ aggr,
                                                  const float* __restrict__ deg,
                                                  const _Float16* __restrict__ w2_p,
                                                  const float* __restrict__ msg_b2,
                                                  const _Float16* __restrict__ u1_p,
                                                  const float* __restrict__ upd_b1,
                                                  const _Float16* __restrict__ u2_p,
                                                  const float* __restrict__ upd_b2) {
  __shared__ __align__(16) _Float16 sAgg[16 * HID];
  __shared__ __align__(16) _Float16 sX[16 * HID];
  __shared__ __align__(16) _Float16 sM2[16 * HID];
  __shared__ __align__(16) _Float16 sG[16 * HID];
  __shared__ float sDg[16];
  int tid = threadIdx.x, wave = tid >> 5, lane = tid & 31;
  int rbase = blockIdx.x * 16;
  for (int idx = tid; idx < 16 * HID; idx += 256) {
    sAgg[idx] = (_Float16)aggr[rbase * HID + idx];
    sX[idx] = xh[rbase * HID + idx];
  }
  if (tid < 16) sDg[tid] = deg[rbase + tid];
  __syncthreads();
  int col = wave * 16 + (lane & 15);
  int mbase = (lane < 16) ? 0 : 8;

  // m2 = aggr @ msg_w2 + deg*b2
  v8f acc = {};
#pragma unroll
  for (int ks = 0; ks < 4; ++ks)
    acc = wmma_f16(load_A(sAgg, lane, HID, ks * 32),
                   load_Bp(w2_p, wave * 4 + ks, lane), acc);
  float b2v = msg_b2[col];
#pragma unroll
  for (int v = 0; v < 8; ++v)
    sM2[(mbase + v) * HID + col] = (_Float16)(acc[v] + sDg[mbase + v] * b2v);
  __syncthreads();

  // g = silu([x | m2] @ upd_w1 + b1)   (K = 256: x-half ks 0..3, m2-half ks 4..7)
  v8f acc2 = {};
#pragma unroll
  for (int ks = 0; ks < 4; ++ks)
    acc2 = wmma_f16(load_A(sX, lane, HID, ks * 32),
                    load_Bp(u1_p, wave * 8 + ks, lane), acc2);
#pragma unroll
  for (int ks = 0; ks < 4; ++ks)
    acc2 = wmma_f16(load_A(sM2, lane, HID, ks * 32),
                    load_Bp(u1_p, wave * 8 + 4 + ks, lane), acc2);
  float b1v = upd_b1[col];
#pragma unroll
  for (int v = 0; v < 8; ++v)
    sG[(mbase + v) * HID + col] = (_Float16)silu(acc2[v] + b1v);
  __syncthreads();

  // x += g @ upd_w2 + b2
  v8f acc3 = {};
#pragma unroll
  for (int ks = 0; ks < 4; ++ks)
    acc3 = wmma_f16(load_A(sG, lane, HID, ks * 32),
                    load_Bp(u2_p, wave * 4 + ks, lane), acc3);
  float b2u = upd_b2[col];
#pragma unroll
  for (int v = 0; v < 8; ++v) {
    int row = rbase + mbase + v;
    float xn = x[row * HID + col] + acc3[v] + b2u;
    x[row * HID + col] = xn;
    xh[row * HID + col] = (_Float16)xn;
  }
}

// ---------------------------------------------------------------------------
// Kernel 5: segment-mean pooling + 128->64->1 MLP (tiny; single workgroup)
// ---------------------------------------------------------------------------
__global__ __launch_bounds__(256) void readout_kernel(const float* __restrict__ x,
                                                      const int* __restrict__ batch,
                                                      const float* __restrict__ w1,
                                                      const float* __restrict__ b1,
                                                      const float* __restrict__ w2,
                                                      const float* __restrict__ b2,
                                                      float* __restrict__ out) {
  __shared__ float sums[NMOLS * HID];
  __shared__ float cnt[NMOLS];
  __shared__ float sH[NMOLS * 64];
  int tid = threadIdx.x;
  for (int idx = tid; idx < NMOLS * HID; idx += 256) sums[idx] = 0.0f;
  if (tid < NMOLS) cnt[tid] = 0.0f;
  __syncthreads();
  if (tid < HID) { // thread owns one feature column -> no races
    for (int a = 0; a < NA; ++a) sums[batch[a] * HID + tid] += x[a * HID + tid];
  } else if (tid < HID + NMOLS) {
    int m = tid - HID; float c = 0.0f;
    for (int a = 0; a < NA; ++a) if (batch[a] == m) c += 1.0f;
    cnt[m] = c;
  }
  __syncthreads();
  for (int idx = tid; idx < NMOLS * 64; idx += 256) {
    int m = idx >> 6, k = idx & 63;
    float inv = 1.0f / fmaxf(cnt[m], 1.0f);
    float a = b1[k];
    for (int c = 0; c < HID; ++c) a += (sums[m * HID + c] * inv) * w1[c * 64 + k];
    sH[idx] = silu(a);
  }
  __syncthreads();
  if (tid < NMOLS) {
    float a = b2[0];
    for (int k = 0; k < 64; ++k) a += sH[tid * 64 + k] * w2[k];
    out[tid] = a;
  }
}

// ---------------------------------------------------------------------------
extern "C" void kernel_launch(void* const* d_in, const int* in_sizes, int n_in,
                              void* d_out, int out_size, void* d_ws, size_t ws_size,
                              hipStream_t stream) {
  (void)in_sizes; (void)n_in; (void)out_size; (void)ws_size;
  const int*   an      = (const int*)d_in[0];
  const float* pos     = (const float*)d_in[1];
  const int*   batch   = (const int*)d_in[2];
  const float* emb     = (const float*)d_in[3];
  const float* centers = (const float*)d_in[4];
  const float* widths  = (const float*)d_in[5];
  const float* msg_w1  = (const float*)d_in[6];
  const float* msg_b1  = (const float*)d_in[7];
  const float* msg_w2  = (const float*)d_in[8];
  const float* msg_b2  = (const float*)d_in[9];
  const float* upd_w1  = (const float*)d_in[10];
  const float* upd_b1  = (const float*)d_in[11];
  const float* upd_w2  = (const float*)d_in[12];
  const float* upd_b2  = (const float*)d_in[13];
  const float* out_w1  = (const float*)d_in[14];
  const float* out_b1  = (const float*)d_in[15];
  const float* out_w2  = (const float*)d_in[16];
  const float* out_b2  = (const float*)d_in[17];

  char* ws = (char*)d_ws;
  float*     x     = (float*)(ws + 0);            // 262144
  _Float16*  xh    = (_Float16*)(ws + 262144);    // 131072
  float*     T     = (float*)(ws + 393216);       // 262144
  float*     aggr  = (float*)(ws + 655360);       // 262144
  float*     deg   = (float*)(ws + 917504);       // 2048
  _Float16*  w1xp  = (_Float16*)(ws + 919552);    // 131072
  _Float16*  w1rp  = (_Float16*)(ws + 1050624);   // 65536
  _Float16*  w2p   = (_Float16*)(ws + 1116160);   // 131072
  _Float16*  u1p   = (_Float16*)(ws + 1247232);   // 262144
  _Float16*  u2p   = (_Float16*)(ws + 1509376);   // 131072
  float*     aggrP = (float*)(ws + 1640448);      // NCHUNK*512*128*4 = 2097152
  float*     degP  = (float*)(ws + 3737600);      // NCHUNK*512*4     = 16384

  prep_kernel<<<208, 256, 0, stream>>>(an, emb, msg_w1, msg_w2, upd_w1, upd_w2,
                                       x, xh, w1xp, w1rp, w2p, u1p, u2p);
  for (int b = 0; b < NB; ++b) {
    tgemm_kernel<<<32, 256, 0, stream>>>(xh, w1xp + b * HID * HID,
                                         msg_b1 + b * HID, T);
    msg_kernel<<<dim3(32, NCHUNK), 256, 0, stream>>>(pos, centers, widths, T,
                                                     w1rp + b * 64 * HID,
                                                     aggrP, degP);
    reduce_kernel<<<64, 256, 0, stream>>>(aggrP, degP, aggr, deg);
    upd_kernel<<<32, 256, 0, stream>>>(x, xh, aggr, deg,
                                       w2p + b * HID * HID, msg_b2 + b * HID,
                                       u1p + b * 2 * HID * HID, upd_b1 + b * HID,
                                       u2p + b * HID * HID, upd_b2 + b * HID);
  }
  readout_kernel<<<1, 256, 0, stream>>>(x, batch, out_w1, out_b1, out_w2, out_b2,
                                        (float*)d_out);
}
